// LSM_12627203850874
// MI455X (gfx1250) — matrix operand order repaired and checked by reading, not compiled, not run on hardware
//
#include <hip/hip_runtime.h>

typedef __attribute__((ext_vector_type(2)))  float  v2f;
typedef __attribute__((ext_vector_type(8)))  float  v8f;
typedef __attribute__((ext_vector_type(8)))  __bf16 v8bf;
typedef __attribute__((ext_vector_type(16))) __bf16 v16bf;

#define T_STEPS 256
#define B_DIM   64
#define IN_DIM  1024
#define N_DIM   2048
#define KB_CNT  (N_DIM / 32)   /* 64 K-blocks of 32 */
#define NT_CNT  (N_DIM / 16)   /* 128 N tiles */
#define ALPHA_C 0.9f
#define BETA_C  0.9f
#define TH_C    20.0f

__device__ __forceinline__ unsigned int pack2(__bf16 a, __bf16 b) {
  unsigned short ua = __builtin_bit_cast(unsigned short, a);
  unsigned short ub = __builtin_bit_cast(unsigned short, b);
  return (unsigned int)ua | ((unsigned int)ub << 16);
}

// ---------------------------------------------------------------------------
// LDS-tiled transpose (for WinT): out[c*rows + r] = in[r*cols + c]
// ---------------------------------------------------------------------------
__global__ void lsm_transpose(const float* __restrict__ in, float* __restrict__ out,
                              int rows, int cols) {
  __shared__ float tile[32][33];
  const int bx = blockIdx.x * 32;
  const int by = blockIdx.y * 32;
#pragma unroll
  for (int j = 0; j < 32; j += 8)
    tile[threadIdx.y + j][threadIdx.x] =
        in[(size_t)(by + threadIdx.y + j) * cols + bx + threadIdx.x];
  __syncthreads();
#pragma unroll
  for (int j = 0; j < 32; j += 8)
    out[(size_t)(bx + threadIdx.y + j) * rows + by + threadIdx.x] =
        tile[threadIdx.x][threadIdx.y + j];
}

// ---------------------------------------------------------------------------
// Pack Wlsm into bf16 hi/lo B-matrix fragments laid out as this code's
// 32x16 bf16 B VGPR image (contiguous k, matching A):  frag[((nt*KB + kb)*32 + lane)*8 + j] packs
// K = kb*32 + 16*(lane>=16) + 2j .. +1  at column n = nt*16 + (lane&15).
// B[k][n] = Wlsm[n][k], read directly from row-major Wlsm.
// ---------------------------------------------------------------------------
__global__ void lsm_pack_b(const float* __restrict__ Wlsm,
                           unsigned int* __restrict__ hi,
                           unsigned int* __restrict__ lo) {
  const int idx  = blockIdx.x * 256 + threadIdx.x;   // 2^21 dword slots
  const int j    = idx & 7;
  const int lane = (idx >> 3) & 31;
  const int kb   = (idx >> 8) & (KB_CNT - 1);
  const int nt   = idx >> 14;
  const int n = nt * 16 + (lane & 15);
  const int k = kb * 32 + ((lane >> 4) << 4) + 2 * j;
  const float w0 = Wlsm[(size_t)n * N_DIM + k];
  const float w1 = Wlsm[(size_t)n * N_DIM + k + 1];
  const __bf16 h0 = (__bf16)w0, h1 = (__bf16)w1;
  const __bf16 l0 = (__bf16)(w0 - (float)h0);
  const __bf16 l1 = (__bf16)(w1 - (float)h1);
  hi[idx] = pack2(h0, h1);
  lo[idx] = pack2(l0, l1);
}

// ---------------------------------------------------------------------------
// cur_all = x @ WinT + b1 (into d_out), exact fp32 WMMA. 1 wave per 16x16 tile.
// ---------------------------------------------------------------------------
__global__ void lsm_input_gemm(const float* __restrict__ A,   // [M x IN_DIM]
                               const float* __restrict__ Bt,  // [IN_DIM x N_DIM]
                               const float* __restrict__ bias,
                               float* __restrict__ C) {
  const int lane = threadIdx.x & 31;
  const int wave = threadIdx.x >> 5;
  const int tile = blockIdx.x * 8 + wave;
  const int mt = tile >> 7;
  const int nt = tile & 127;
  const int m0 = mt * 16, n0 = nt * 16;
  const int half = lane >> 4;
  const int lid  = lane & 15;

  const float* aRow = A  + (size_t)(m0 + lid) * IN_DIM + 2 * half;
  const float* bCol = Bt + (size_t)(2 * half) * N_DIM + n0 + lid;

  v8f c = {0.f, 0.f, 0.f, 0.f, 0.f, 0.f, 0.f, 0.f};
#pragma unroll 4
  for (int k = 0; k < IN_DIM; k += 4) {
    v2f a = *(const v2f*)(aRow + k);
    v2f b;
    b[0] = bCol[(size_t)(k)     * N_DIM];
    b[1] = bCol[(size_t)(k + 1) * N_DIM];
    c = __builtin_amdgcn_wmma_f32_16x16x4_f32(false, a, false, b, (short)0, c,
                                              false, false);
  }
  const float bs = bias[n0 + lid];
  float* cp = C + (size_t)(m0 + 8 * half) * N_DIM + n0 + lid;
#pragma unroll
  for (int r = 0; r < 8; ++r) cp[(size_t)r * N_DIM] = c[r] + bs;
}

// ---------------------------------------------------------------------------
// One LSM step.  rec = spk_prev @ Wlsm^T via bf16 hi/lo WMMA (exact: spk is
// binary).  One block per 16x16 tile, split-K over 8 waves + LDS reduction.
// curspk holds cur[t] on entry, overwritten with fp32 spikes; bf16 spikes go
// to spkOutB for the next step's A-matrix.
// ---------------------------------------------------------------------------
__global__ void lsm_step(const __bf16* __restrict__ spkPrevB, // [B x N] bf16 or null
                         const unsigned int* __restrict__ bHi,
                         const unsigned int* __restrict__ bLo,
                         const float* __restrict__ brec,
                         float* __restrict__ syn,
                         float* __restrict__ mem,
                         float* curspk,
                         __bf16* __restrict__ spkOutB) {
  __shared__ float red[8][256];
  const int lane = threadIdx.x & 31;
  const int wave = threadIdx.x >> 5;
  const int mt = blockIdx.x >> 7;
  const int nt = blockIdx.x & 127;
  const int m0 = mt * 16, n0 = nt * 16;
  const int half = lane >> 4;
  const int lid  = lane & 15;

  v8f c = {0.f, 0.f, 0.f, 0.f, 0.f, 0.f, 0.f, 0.f};
  if (spkPrevB != nullptr) {
    const int kbBeg = wave * (KB_CNT / 8);   // 8 K-blocks of 32 per wave
    const __bf16* aBase = spkPrevB + (size_t)(m0 + lid) * N_DIM + 16 * half;
    const unsigned int* fragBase = bHi; // indexed below
#pragma unroll 2
    for (int kb = 0; kb < KB_CNT / 8; ++kb) {
      const int kbg = kbBeg + kb;
      const __bf16* ap = aBase + kbg * 32;
      v8bf a0 = *(const v8bf*)(ap);        // K = 32kbg + 16h .. +7
      v8bf a1 = *(const v8bf*)(ap + 8);    // K = 32kbg + 16h + 8 .. +7
      v16bf a = __builtin_shufflevector(a0, a1, 0, 1, 2, 3, 4, 5, 6, 7,
                                        8, 9, 10, 11, 12, 13, 14, 15);
      const size_t fidx = ((size_t)(nt * KB_CNT + kbg) * 32 + lane) * 8;
      v16bf bh = *(const v16bf*)(fragBase + fidx);
      v16bf bl = *(const v16bf*)(bLo + fidx);
      c = __builtin_amdgcn_wmma_f32_16x16x32_bf16(false, a, false, bh, (short)0,
                                                  c, false, false);
      c = __builtin_amdgcn_wmma_f32_16x16x32_bf16(false, a, false, bl, (short)0,
                                                  c, false, false);
    }
  }
  // stash fragment: element (lane, r) at red[wave][r*32 + lane] (conflict-free)
#pragma unroll
  for (int r = 0; r < 8; ++r) red[wave][r * 32 + lane] = c[r];
  __syncthreads();

  // each of the 256 threads reduces + finishes one C element
  const int e  = threadIdx.x;
  float acc = 0.f;
#pragma unroll
  for (int w = 0; w < 8; ++w) acc += red[w][e];
  const int r  = e >> 5;
  const int el = e & 31;
  const int m  = m0 + r + 8 * (el >> 4);
  const int n  = n0 + (el & 15);
  const size_t off = (size_t)m * N_DIM + n;

  const float inp   = curspk[off] + acc + brec[n];
  const float mprev = mem[off];
  const float s     = ALPHA_C * syn[off] + inp;
  const float reset = (mprev > TH_C) ? TH_C : 0.0f;
  const float mnew  = BETA_C * mprev + s - reset;
  const float spk   = (mnew > TH_C) ? 1.0f : 0.0f;
  syn[off]     = s;
  mem[off]     = mnew;
  curspk[off]  = spk;
  spkOutB[off] = (__bf16)spk;   // exact
}

__global__ void lsm_zero(float* __restrict__ p, int n) {
  int i = blockIdx.x * blockDim.x + threadIdx.x;
  if (i < n) p[i] = 0.0f;
}

// ---------------------------------------------------------------------------
extern "C" void kernel_launch(void* const* d_in, const int* in_sizes, int n_in,
                              void* d_out, int out_size, void* d_ws, size_t ws_size,
                              hipStream_t stream) {
  (void)in_sizes; (void)n_in; (void)out_size; (void)ws_size;
  const float* x    = (const float*)d_in[0];  // [T,B,IN]
  const float* Win  = (const float*)d_in[1];  // [N,IN]
  const float* b1   = (const float*)d_in[2];  // [N]
  const float* Wlsm = (const float*)d_in[3];  // [N,N]
  const float* brec = (const float*)d_in[4];  // [N]
  float* out = (float*)d_out;                 // [T,B,N]

  // workspace: bHi | bLo | WinT | syn | mem | spkB0 | spkB1  (~25.5 MB)
  const size_t FRAG_DW = (size_t)NT_CNT * KB_CNT * 32 * 8;  // 2,097,152 dwords
  char* ws = (char*)d_ws;
  unsigned int* bHi = (unsigned int*)ws;
  unsigned int* bLo = bHi + FRAG_DW;
  float* WinT = (float*)(bLo + FRAG_DW);
  float* syn  = WinT + (size_t)IN_DIM * N_DIM;
  float* memv = syn + (size_t)B_DIM * N_DIM;
  __bf16* spkB0 = (__bf16*)(memv + (size_t)B_DIM * N_DIM);
  __bf16* spkB1 = spkB0 + (size_t)B_DIM * N_DIM;

  dim3 tb(32, 8);
  // WinT[k*N + n] = Win[n*IN + k]
  lsm_transpose<<<dim3(IN_DIM / 32, N_DIM / 32), tb, 0, stream>>>(Win, WinT, N_DIM, IN_DIM);
  // pack Wlsm -> bf16 hi/lo B fragments
  lsm_pack_b<<<(int)(FRAG_DW / 256), 256, 0, stream>>>(Wlsm, bHi, bLo);

  // cur_all = x @ WinT + b1 -> d_out
  const int M = T_STEPS * B_DIM;
  const int tiles = (M / 16) * (N_DIM / 16);
  lsm_input_gemm<<<tiles / 8, 256, 0, stream>>>(x, WinT, b1, out);

  // zero syn and mem (contiguous) every call -> replay-deterministic
  lsm_zero<<<(2 * B_DIM * N_DIM + 255) / 256, 256, 0, stream>>>(syn, 2 * B_DIM * N_DIM);

  __bf16* spkbuf[2] = {spkB0, spkB1};
  for (int t = 0; t < T_STEPS; ++t) {
    const __bf16* readB = (t == 0) ? nullptr : spkbuf[(t + 1) & 1];
    __bf16* writeB = spkbuf[t & 1];
    float* slab = out + (size_t)t * B_DIM * N_DIM;
    lsm_step<<<(B_DIM / 16) * (N_DIM / 16), 256, 0, stream>>>(
        readB, bHi, bLo, brec, syn, memv, slab, writeB);
  }
}
